// VQVAE375to291_61177514164811
// MI455X (gfx1250) — matrix operand order repaired and checked by reading, not compiled, not run on hardware
//
#include <hip/hip_runtime.h>
#include <hip/hip_bf16.h>

// ---------------------------------------------------------------------------
// VQ-VAE forward for MI455X (gfx1250, wave32, WMMA).
// All GEMMs: v_wmma_f32_16x16x32_f16, f32 accumulate, f32->f16 in loaders.
// Shapes are padded once (K=375->384, N=291->304) so GEMM inner loops are
// branchless straight-line loads (b128 for A, immediate-offset b32 for B).
// ---------------------------------------------------------------------------

#define BB  32
#define TT  512
#define INW 375
#define INP 384          // padded input K
#define DD  512
#define HH  8
#define HDD 64
#define LL  6
#define FFNW 1024
#define LATW 256
#define CDW 64
#define KC  512
#define DLY 4
#define DHW 512
#define DFFW 1024
#define OUTW 291
#define OUTP 304         // padded output N
#define MTOK (BB * TT)   // 16384 rows everywhere

typedef __attribute__((ext_vector_type(16))) _Float16 v16h;
typedef __attribute__((ext_vector_type(8)))  float    v8f;

__device__ __forceinline__ v8f wmma_f16(v16h a, v16h b, v8f c) {
  return __builtin_amdgcn_wmma_f32_16x16x32_f16(
      false, a, false, b, (short)0, c, false, false);
}

__device__ __forceinline__ float gelu_exact(float x) {
  return 0.5f * x * (1.0f + erff(x * 0.70710678118654752f));
}

// ---- unguarded fragment loaders -------------------------------------------
// A (16x32 f16): lane<16 -> M=lane, K 0..15; lane>=16 -> M=lane-16,
// K 16..31.  Rows contiguous stride LDA -> four b128 loads.
template<int LDA>
__device__ __forceinline__ v16h load_a_ng(const float* __restrict__ A,
                                          int m0, int k0, int lane,
                                          float scale = 1.0f) {
  const float* row = A + (long)(m0 + (lane & 15)) * LDA + k0 + ((lane >> 4) * 16);
  float4 u0 = *reinterpret_cast<const float4*>(row);
  float4 u1 = *reinterpret_cast<const float4*>(row + 4);
  float4 u2 = *reinterpret_cast<const float4*>(row + 8);
  float4 u3 = *reinterpret_cast<const float4*>(row + 12);
  v16h a;
  a[0]=(_Float16)(u0.x*scale); a[1]=(_Float16)(u0.y*scale);
  a[2]=(_Float16)(u0.z*scale); a[3]=(_Float16)(u0.w*scale);
  a[4]=(_Float16)(u1.x*scale); a[5]=(_Float16)(u1.y*scale);
  a[6]=(_Float16)(u1.z*scale); a[7]=(_Float16)(u1.w*scale);
  a[8]=(_Float16)(u2.x*scale); a[9]=(_Float16)(u2.y*scale);
  a[10]=(_Float16)(u2.z*scale); a[11]=(_Float16)(u2.w*scale);
  a[12]=(_Float16)(u3.x*scale); a[13]=(_Float16)(u3.y*scale);
  a[14]=(_Float16)(u3.z*scale); a[15]=(_Float16)(u3.w*scale);
  return a;
}

// B (32x16 f16) from row-major W[*, LDB]: lane&15 = N column; lane half
// selects K 0..15 / 16..31.  LDB compile-time -> 16 immediate-offset b32.
template<int LDB>
__device__ __forceinline__ v16h load_b_ng(const float* __restrict__ Bm,
                                          int k0, int n0, int lane) {
  const float* p = Bm + (long)(k0 + ((lane >> 4) * 16)) * LDB + n0 + (lane & 15);
  v16h b;
#pragma unroll
  for (int j = 0; j < 16; ++j) b[j] = (_Float16)p[(long)j * LDB];
  return b;
}

// ---------------------------------------------------------------------------
// Generic GEMM: Out[M,N] = act(A[M,LDA] @ W[LDA,LDB] + bias) (+ residual)
// Block 256 = 8 waves; wave w -> cols n0 = bx*128 + w*16, rows m0 = by*64.
// 4 accumulators along M: one B fragment feeds 4 WMMAs.
// ---------------------------------------------------------------------------
template<int LDA, int LDB, int ACT, bool HASRES>
__global__ __launch_bounds__(256) void k_linear(
    const float* __restrict__ A, const float* __restrict__ W,
    const float* __restrict__ bias, const float* residual,
    float* Out, int N) {
  int lane = threadIdx.x & 31;
  int wave = threadIdx.x >> 5;
  int n0 = blockIdx.x * 128 + wave * 16;
  int m0 = blockIdx.y * 64;
  if (n0 >= LDB) return;                 // wave-uniform

  v8f c0 = {}, c1 = {}, c2 = {}, c3 = {};
  for (int k0 = 0; k0 < LDA; k0 += 32) {
    v16h b  = load_b_ng<LDB>(W, k0, n0, lane);
    v16h a0 = load_a_ng<LDA>(A, m0,      k0, lane);
    v16h a1 = load_a_ng<LDA>(A, m0 + 16, k0, lane);
    v16h a2 = load_a_ng<LDA>(A, m0 + 32, k0, lane);
    v16h a3 = load_a_ng<LDA>(A, m0 + 48, k0, lane);
    c0 = wmma_f16(a0, b, c0);
    c1 = wmma_f16(a1, b, c1);
    c2 = wmma_f16(a2, b, c2);
    c3 = wmma_f16(a3, b, c3);
  }

  int nc = n0 + (lane & 15);
  if (nc >= N) return;                   // per-lane, but all WMMAs are done
  float bv = bias[nc];
  int mg = m0 + ((lane >> 4) * 8);
#pragma unroll
  for (int mt = 0; mt < 4; ++mt) {
    v8f c = (mt == 0) ? c0 : (mt == 1) ? c1 : (mt == 2) ? c2 : c3;
#pragma unroll
    for (int r = 0; r < 8; ++r) {
      float v = c[r] + bv;
      if (ACT == 1) v = gelu_exact(v);
      long idx = (long)(mg + mt * 16 + r) * N + nc;
      if (HASRES) v += residual[idx];
      Out[idx] = v;
    }
  }
}

// ---------------------------------------------------------------------------
// Zero-pad copy: dst[Rp x Cp] <- src[R x C], zeros elsewhere.
// ---------------------------------------------------------------------------
__global__ __launch_bounds__(256) void k_pad2d(const float* __restrict__ src,
                                               float* __restrict__ dst,
                                               int R, int C, int Cp, long total) {
  long i = (long)blockIdx.x * 256 + threadIdx.x;
  if (i >= total) return;
  long r = i / Cp;
  int c = (int)(i - r * Cp);
  dst[i] = (c < C && r < R) ? src[r * (long)C + c] : 0.0f;
}

// ---------------------------------------------------------------------------
// LayerNorm: one wave per row (D=512 -> 16 elems/lane).
// ---------------------------------------------------------------------------
__global__ __launch_bounds__(256) void k_layernorm(
    const float* __restrict__ X, const float* __restrict__ s,
    const float* __restrict__ b, float* __restrict__ Y, int D) {
  int lane = threadIdx.x & 31;
  int row = blockIdx.x * 8 + (threadIdx.x >> 5);
  const float* x = X + (long)row * D;
  float sum = 0.0f, sq = 0.0f;
  for (int i = lane; i < D; i += 32) { float v = x[i]; sum += v; sq += v * v; }
#pragma unroll
  for (int m = 16; m; m >>= 1) {
    sum += __shfl_xor(sum, m, 32);
    sq  += __shfl_xor(sq,  m, 32);
  }
  float mean = sum / (float)D;
  float var  = sq / (float)D - mean * mean;
  float inv  = rsqrtf(var + 1e-5f);
  float* y = Y + (long)row * D;
  for (int i = lane; i < D; i += 32)
    y[i] = (x[i] - mean) * inv * s[i] + b[i];
}

// ---------------------------------------------------------------------------
// RoPE in-place on the q,k slices of qkv [M, 1536].
// ---------------------------------------------------------------------------
__global__ __launch_bounds__(256) void k_rope(float* __restrict__ qkv) {
  int tid = blockIdx.x * 256 + threadIdx.x;
  int j    = tid & 31;
  int hh   = (tid >> 5) & 7;
  int part = (tid >> 8) & 1;
  int bt   = tid >> 9;
  if (bt >= MTOK) return;
  int t = bt & (TT - 1);
  float freq = __expf(-(float)(2 * j) * (1.0f / (float)HDD) * 9.2103403720f);
  float ang = (float)t * freq;
  float sn, cs;
  __sincosf(ang, &sn, &cs);
  float* p = qkv + (long)bt * (3 * DD) + part * DD + hh * HDD + 2 * j;
  float xe = p[0], xo = p[1];
  p[0] = xe * cs - xo * sn;
  p[1] = xe * sn + xo * cs;
}

// ---------------------------------------------------------------------------
// Flash attention, WMMA.  Grid (B*H, T/64), block 128 (4 waves);
// wave = 16 query rows, streams 32 keys/iter.
// ---------------------------------------------------------------------------
__device__ __forceinline__ v16h load_b_kt(const float* __restrict__ base,
                                          int n0, int k0, int lane) {
  // B[kk][n] = K[key n][feat kk]; 16 contiguous floats -> 4 x b128
  const float* p = base + (long)(n0 + (lane & 15)) * (3 * DD)
                        + k0 + ((lane >> 4) * 16);
  float4 u0 = *reinterpret_cast<const float4*>(p);
  float4 u1 = *reinterpret_cast<const float4*>(p + 4);
  float4 u2 = *reinterpret_cast<const float4*>(p + 8);
  float4 u3 = *reinterpret_cast<const float4*>(p + 12);
  v16h b;
  b[0]=(_Float16)u0.x; b[1]=(_Float16)u0.y; b[2]=(_Float16)u0.z; b[3]=(_Float16)u0.w;
  b[4]=(_Float16)u1.x; b[5]=(_Float16)u1.y; b[6]=(_Float16)u1.z; b[7]=(_Float16)u1.w;
  b[8]=(_Float16)u2.x; b[9]=(_Float16)u2.y; b[10]=(_Float16)u2.z; b[11]=(_Float16)u2.w;
  b[12]=(_Float16)u3.x; b[13]=(_Float16)u3.y; b[14]=(_Float16)u3.z; b[15]=(_Float16)u3.w;
  return b;
}

__device__ __forceinline__ v16h load_b_v(const float* __restrict__ base,
                                         int k0, int n0, int lane) {
  const float* p = base + (long)(k0 + ((lane >> 4) * 16)) * (3 * DD)
                        + n0 + (lane & 15);
  v16h b;
#pragma unroll
  for (int j = 0; j < 16; ++j) b[j] = (_Float16)p[(long)j * (3 * DD)];
  return b;
}

__global__ __launch_bounds__(128) void k_attn(const float* __restrict__ qkv,
                                              float* __restrict__ out) {
  __shared__ _Float16 plds[4][16 * 32];
  int lane = threadIdx.x & 31;
  int wave = threadIdx.x >> 5;
  int bh = blockIdx.x;
  int b = bh >> 3, h = bh & 7;
  int q0 = blockIdx.y * 64 + wave * 16;

  const float* Qb = qkv + (long)b * TT * (3 * DD) + h * HDD;   // part 0
  const float* Kb = Qb + DD;                                   // part 1
  const float* Vb = Qb + 2 * DD;                               // part 2

  v16h qa0 = load_a_ng<3 * DD>(Qb, q0, 0,  lane, 0.125f);      // 1/sqrt(64)
  v16h qa1 = load_a_ng<3 * DD>(Qb, q0, 32, lane, 0.125f);

  float mi[8], li[8];
#pragma unroll
  for (int r = 0; r < 8; ++r) { mi[r] = -1e30f; li[r] = 0.0f; }
  v8f o0 = {}, o1 = {}, o2 = {}, o3 = {};
  _Float16* pl = plds[wave];

  for (int kt = 0; kt < TT; kt += 32) {
    v8f s0 = {}, s1 = {};
    s0 = wmma_f16(qa0, load_b_kt(Kb, kt,      0,  lane), s0);
    s0 = wmma_f16(qa1, load_b_kt(Kb, kt,      32, lane), s0);
    s1 = wmma_f16(qa0, load_b_kt(Kb, kt + 16, 0,  lane), s1);
    s1 = wmma_f16(qa1, load_b_kt(Kb, kt + 16, 32, lane), s1);

    int rowbase = (lane >> 4) * 8;
    int col = lane & 15;
#pragma unroll
    for (int r = 0; r < 8; ++r) {
      float mx = fmaxf(s0[r], s1[r]);
#pragma unroll
      for (int msk = 8; msk; msk >>= 1) mx = fmaxf(mx, __shfl_xor(mx, msk, 32));
      float mn = fmaxf(mi[r], mx);
      float al = __expf(mi[r] - mn);
      float p0 = __expf(s0[r] - mn);
      float p1 = __expf(s1[r] - mn);
      float sm = p0 + p1;
#pragma unroll
      for (int msk = 8; msk; msk >>= 1) sm += __shfl_xor(sm, msk, 32);
      li[r] = li[r] * al + sm;
      mi[r] = mn;
      pl[(rowbase + r) * 32 + col]      = (_Float16)p0;
      pl[(rowbase + r) * 32 + 16 + col] = (_Float16)p1;
      o0[r] *= al; o1[r] *= al; o2[r] *= al; o3[r] *= al;
    }

    // C-layout -> A-layout reload of P (same-wave LDS RAW; DS pipe in-order)
    v16h pa;
    {
      int mr = lane & 15;
      int kh = (lane >> 4) * 16;
#pragma unroll
      for (int i = 0; i < 8; ++i) {
        pa[i]     = pl[mr * 32 + kh + i];
        pa[8 + i] = pl[mr * 32 + kh + 8 + i];
      }
    }
    o0 = wmma_f16(pa, load_b_v(Vb, kt, 0,  lane), o0);
    o1 = wmma_f16(pa, load_b_v(Vb, kt, 16, lane), o1);
    o2 = wmma_f16(pa, load_b_v(Vb, kt, 32, lane), o2);
    o3 = wmma_f16(pa, load_b_v(Vb, kt, 48, lane), o3);
  }

  int row0 = q0 + (lane >> 4) * 8;
  int nc = lane & 15;
#pragma unroll
  for (int r = 0; r < 8; ++r) {
    float inv = 1.0f / li[r];
    long base = ((long)b * TT + row0 + r) * DD + h * HDD + nc;
    out[base]      = o0[r] * inv;
    out[base + 16] = o1[r] * inv;
    out[base + 32] = o2[r] * inv;
    out[base + 48] = o3[r] * inv;
  }
}

// ---------------------------------------------------------------------------
// VQ: per 64-d sub-vector, argmin over 512 codes; output = chosen codeword.
// ---------------------------------------------------------------------------
__global__ __launch_bounds__(256) void k_vq(const float* __restrict__ Z,
                                            const float* __restrict__ cb,
                                            float* __restrict__ Q) {
  __shared__ float zsh[8][CDW];
  int lane = threadIdx.x & 31;
  int wave = threadIdx.x >> 5;
  long s = (long)blockIdx.x * 8 + wave;
  const float* z = Z + s * CDW;
  zsh[wave][lane]      = z[lane];
  zsh[wave][lane + 32] = z[lane + 32];
  __syncthreads();

  float best = 3.4e38f;
  int bidx = 0x7fffffff;
  for (int c = lane; c < KC; c += 32) {
    const float* cw = cb + (long)c * CDW;
    float d = 0.0f;
#pragma unroll 8
    for (int k = 0; k < CDW; ++k) {
      float t = zsh[wave][k] - cw[k];
      d = fmaf(t, t, d);
    }
    if (d < best) { best = d; bidx = c; }
  }
#pragma unroll
  for (int msk = 16; msk; msk >>= 1) {
    float ob = __shfl_xor(best, msk, 32);
    int   oi = __shfl_xor(bidx, msk, 32);
    if (ob < best || (ob == best && oi < bidx)) { best = ob; bidx = oi; }
  }
  const float* cw = cb + (long)bidx * CDW;
  float* q = Q + s * CDW;
  q[lane]      = cw[lane];
  q[lane + 32] = cw[lane + 32];
}

// ---------------------------------------------------------------------------
// Host-side orchestration
// ---------------------------------------------------------------------------
template<int LDA, int LDB, int ACT, bool HASRES>
static inline void launch_lin(const float* A, const float* W, const float* bias,
                              const float* res, float* Out, int N,
                              hipStream_t s) {
  dim3 grid((LDB + 127) / 128, MTOK / 64);
  k_linear<LDA, LDB, ACT, HASRES><<<grid, 256, 0, s>>>(A, W, bias, res, Out, N);
}

extern "C" void kernel_launch(void* const* d_in, const int* in_sizes, int n_in,
                              void* d_out, int out_size, void* d_ws, size_t ws_size,
                              hipStream_t stream) {
  const float* x      = (const float*)d_in[0];
  const float* in_w   = (const float*)d_in[1];
  const float* in_b   = (const float*)d_in[2];
  const float* ln1_s  = (const float*)d_in[3];
  const float* ln1_b  = (const float*)d_in[4];
  const float* qkv_w  = (const float*)d_in[5];
  const float* qkv_b  = (const float*)d_in[6];
  const float* out_w  = (const float*)d_in[7];
  const float* out_b  = (const float*)d_in[8];
  const float* ln2_s  = (const float*)d_in[9];
  const float* ln2_b  = (const float*)d_in[10];
  const float* ff1_w  = (const float*)d_in[11];
  const float* ff1_b  = (const float*)d_in[12];
  const float* ff2_w  = (const float*)d_in[13];
  const float* ff2_b  = (const float*)d_in[14];
  const float* lat_w  = (const float*)d_in[15];
  const float* lat_b  = (const float*)d_in[16];
  const float* cbook  = (const float*)d_in[17];
  const float* unl_w  = (const float*)d_in[18];
  const float* unl_b  = (const float*)d_in[19];
  const float* dln_s  = (const float*)d_in[20];
  const float* dln_b  = (const float*)d_in[21];
  const float* dff1_w = (const float*)d_in[22];
  const float* dff1_b = (const float*)d_in[23];
  const float* dff2_w = (const float*)d_in[24];
  const float* dff2_b = (const float*)d_in[25];
  const float* ow     = (const float*)d_in[26];
  const float* obv    = (const float*)d_in[27];

  // Workspace carve-up (floats), ~295 MB total.
  float* ws    = (float*)d_ws;
  float* h     = ws;                               // [M, 512]
  float* y     = h     + (long)MTOK * DD;          // [M, 512]
  float* qkvb  = y     + (long)MTOK * DD;          // [M, 1536]
  float* attn  = qkvb  + (long)MTOK * (3 * DD);    // [M, 512]
  float* ff    = attn  + (long)MTOK * DD;          // [M, 1024]
  float* z     = ff;                               // [M, 256] (reuses ff)
  float* qz    = ff    + (long)MTOK * LATW;        // [M, 256]
  float* xpad  = ff    + (long)MTOK * DFFW;        // [M, 384]
  float* wpin  = xpad  + (long)MTOK * INP;         // [384, 512]
  float* wpout = wpin  + (long)INP * DD;           // [512, 304]
  float* bpout = wpout + (long)DD * OUTP;          // [304]

  // 0) zero-pad awkward shapes once (K=375 -> 384, N=291 -> 304)
  {
    long t1 = (long)MTOK * INP;
    k_pad2d<<<(t1 + 255) / 256, 256, 0, stream>>>(x, xpad, MTOK, INW, INP, t1);
    long t2 = (long)INP * DD;
    k_pad2d<<<(t2 + 255) / 256, 256, 0, stream>>>(in_w, wpin, INW, DD, DD, t2);
    long t3 = (long)DD * OUTP;
    k_pad2d<<<(t3 + 255) / 256, 256, 0, stream>>>(ow, wpout, DD, OUTW, OUTP, t3);
    long t4 = OUTP;
    k_pad2d<<<(t4 + 255) / 256, 256, 0, stream>>>(obv, bpout, 1, OUTW, OUTP, t4);
  }

  // 1) input projection
  launch_lin<INP, DD, 0, false>(xpad, wpin, in_b, nullptr, h, DD, stream);

  // 2) encoder layers
  for (int i = 0; i < LL; ++i) {
    k_layernorm<<<MTOK / 8, 256, 0, stream>>>(h, ln1_s + i * DD, ln1_b + i * DD, y, DD);
    launch_lin<DD, 3 * DD, 0, false>(y, qkv_w + (long)i * DD * 3 * DD,
                                     qkv_b + (long)i * 3 * DD, nullptr, qkvb,
                                     3 * DD, stream);
    k_rope<<<(MTOK * 512) / 256, 256, 0, stream>>>(qkvb);
    k_attn<<<dim3(BB * HH, TT / 64), 128, 0, stream>>>(qkvb, attn);
    launch_lin<DD, DD, 0, true>(attn, out_w + (long)i * DD * DD,
                                out_b + (long)i * DD, h, h, DD, stream);
    k_layernorm<<<MTOK / 8, 256, 0, stream>>>(h, ln2_s + i * DD, ln2_b + i * DD, y, DD);
    launch_lin<DD, FFNW, 1, false>(y, ff1_w + (long)i * DD * FFNW,
                                   ff1_b + (long)i * FFNW, nullptr, ff, FFNW, stream);
    launch_lin<FFNW, DD, 0, true>(ff, ff2_w + (long)i * FFNW * DD,
                                  ff2_b + (long)i * DD, h, h, DD, stream);
  }

  // 3) latent projection + VQ (straight-through forward == codeword)
  launch_lin<DD, LATW, 0, false>(h, lat_w, lat_b, nullptr, z, LATW, stream);
  k_vq<<<(MTOK * (LATW / CDW)) / 8, 256, 0, stream>>>(z, cbook, qz);
  launch_lin<LATW, DHW, 0, false>(qz, unl_w, unl_b, nullptr, h, DHW, stream);

  // 4) decoder layers
  for (int i = 0; i < DLY; ++i) {
    k_layernorm<<<MTOK / 8, 256, 0, stream>>>(h, dln_s + i * DHW, dln_b + i * DHW, y, DHW);
    launch_lin<DHW, DFFW, 1, false>(y, dff1_w + (long)i * DHW * DFFW,
                                    dff1_b + (long)i * DFFW, nullptr, ff, DFFW, stream);
    launch_lin<DFFW, DHW, 0, true>(ff, dff2_w + (long)i * DFFW * DHW,
                                   dff2_b + (long)i * DHW, h, h, DHW, stream);
  }

  // 5) output projection (stores guarded to N=291, loads use padded 304)
  launch_lin<DD, OUTP, 0, false>(h, wpout, bpout, nullptr, (float*)d_out,
                                 OUTW, stream);
}
